// LongJointRegAndParc_26388279067312
// MI455X (gfx1250) — matrix operand order repaired and checked by reading, not compiled, not run on hardware
//
#include <hip/hip_runtime.h>
#include <hip/hip_bf16.h>

// ---------------------------------------------------------------------------
// Types for CDNA5 WMMA (bf16 A/B, f32 accum)
// ---------------------------------------------------------------------------
typedef __bf16 bf16_t;
typedef __attribute__((ext_vector_type(8)))  __bf16 bf16x8;
typedef __attribute__((ext_vector_type(16))) __bf16 bf16x16;
typedef __attribute__((ext_vector_type(8)))  float  floatx8;

// Exact unsigned division by runtime constant: q = (n*M)>>36, M from host.
// Exact while n*(d-1) < 2^36 (max here ~2^34).
__device__ __forceinline__ unsigned mdiv36(unsigned n, unsigned long long M) {
  return (unsigned)(((unsigned long long)n * M) >> 36);
}

// ---------------------------------------------------------------------------
// im2col + f32->bf16 pack with K zero-padded to Kp (multiple of 32):
//   D[z][n*Kp + k] = (k < K) ? x[z][nbr[7n + k/Cin]*Cin + k%Cin] : 0
// ---------------------------------------------------------------------------
__global__ __launch_bounds__(256) void k_pack_gather(
    const float* __restrict__ S, const int* __restrict__ nbr,
    bf16_t* __restrict__ D, int N, int Cin, int K, int Kp,
    unsigned long long Mkp, unsigned long long Mcin,
    long long sSB, long long dSB)
{
  const long long i = (long long)blockIdx.x * 256 + threadIdx.x;
  if (i >= (long long)N * Kp) return;
  const unsigned n = mdiv36((unsigned)i, Mkp);
  const unsigned k = (unsigned)i - n * (unsigned)Kp;
  float v = 0.f;
  if (k < (unsigned)K) {
    const unsigned j = mdiv36(k, Mcin);
    const unsigned c = k - j * (unsigned)Cin;
    const float* s = S + (long long)blockIdx.z * sSB;
    v = s[(long long)nbr[(long long)n * 7 + j] * Cin + c];
  }
  (D + (long long)blockIdx.z * dSB)[i] = (bf16_t)v;
}

// Dense f32->bf16 pack with zero pad:  D[z][n*Kp + k] = (k<K) ? S[z][n*K+k] : 0
__global__ __launch_bounds__(256) void k_pack_dense(
    const float* __restrict__ S, bf16_t* __restrict__ D,
    int N, int K, int Kp, unsigned long long Mkp,
    long long sSB, long long dSB)
{
  const long long i = (long long)blockIdx.x * 256 + threadIdx.x;
  if (i >= (long long)N * Kp) return;
  const unsigned n = mdiv36((unsigned)i, Mkp);
  const unsigned k = (unsigned)i - n * (unsigned)Kp;
  float v = 0.f;
  if (k < (unsigned)K) v = (S + (long long)blockIdx.z * sSB)[(long long)n * K + k];
  (D + (long long)blockIdx.z * dSB)[i] = (bf16_t)v;
}

// ---------------------------------------------------------------------------
// Dense bf16 GEMM on pre-packed operands:
//   Out[z][m, n] = sum_k A[z][m*Kp+k] * W[n*Kp+k]  (+ bias[n])
// A: [N, Kp] bf16, W: [Cout, Kp] bf16, Kp % 32 == 0 (zero padded).
// One wave = 16(M) x 16(N) tile; per K-step: 4x global_load_b128 + 1 WMMA.
// Row/col clamping instead of masking -> EXEC all-ones, branch-free loop.
// ---------------------------------------------------------------------------
__global__ __launch_bounds__(128) void k_gemm_bf16(
    const bf16_t* __restrict__ A, const bf16_t* __restrict__ W,
    const float* __restrict__ bias, float* __restrict__ Out,
    int N, int Cout, int Kp, int orow,
    long long aSB, long long oSB)
{
  const int wave = threadIdx.x >> 5;
  const int lane = threadIdx.x & 31;
  const int half = lane >> 4;
  const int lm   = lane & 15;
  const int mBase = (blockIdx.x * 4 + wave) * 16;
  const int nBase = blockIdx.y * 16;
  if (mBase >= N) return;                         // wave-uniform exit

  const int mA = min(mBase + lm, N - 1);          // clamped, no masking
  const int nB = min(nBase + lm, Cout - 1);
  const bf16_t* arow = A + (long long)blockIdx.z * aSB + (long long)mA * Kp;
  const bf16_t* brow = W + (long long)nB * Kp;

  floatx8 acc = {};
  for (int k0 = 0; k0 < Kp; k0 += 32) {
    // A frag elements (v,t): k = k0 + (v>=4)*16 + half*8 + (v&3)*2 + t
    //   -> two contiguous 16B chunks per lane
    const bf16x8 a0 = *(const bf16x8*)(arow + k0 + (half << 3));
    const bf16x8 a1 = *(const bf16x8*)(arow + k0 + 16 + (half << 3));
    // B frag elements (v,t): k = k0 + (v>=4)*16 + half*8 + (v&3)*2 + t
    //   -> two contiguous 16B chunks per lane
    const bf16x8 b0 = *(const bf16x8*)(brow + k0 + (half << 3));
    const bf16x8 b1 = *(const bf16x8*)(brow + k0 + 16 + (half << 3));
    const bf16x16 afrag = __builtin_shufflevector(
        a0, a1, 0, 1, 2, 3, 4, 5, 6, 7, 8, 9, 10, 11, 12, 13, 14, 15);
    const bf16x16 bfrag = __builtin_shufflevector(
        b0, b1, 0, 1, 2, 3, 4, 5, 6, 7, 8, 9, 10, 11, 12, 13, 14, 15);
#if defined(__HIP_DEVICE_COMPILE__)
    acc = __builtin_amdgcn_wmma_f32_16x16x32_bf16(
        false, afrag, false, bfrag, (short)0, acc, false, false);
#endif
  }
  float* o = Out + (long long)blockIdx.z * oSB;
#pragma unroll
  for (int r = 0; r < 8; ++r) {
    const int m = mBase + (half << 3) + r;
    const int n = nBase + lm;
    if (m < N && n < Cout) {
      float v = acc[r];
      if (bias) v += bias[n];
      o[(long long)m * orow + n] = v;
    }
  }
}

// ---------------------------------------------------------------------------
// BatchNorm batch statistics (biased var) per (batch z, channel blockIdx.x)
// ---------------------------------------------------------------------------
__global__ __launch_bounds__(256) void k_bn_stats(
    const float* __restrict__ X, float* __restrict__ MV,
    int N, int C, long long xSB, long long mvSB)
{
  const int c = blockIdx.x;
  const float* x = X + (long long)blockIdx.z * xSB;
  float s = 0.f, s2 = 0.f;
  for (int i = threadIdx.x; i < N; i += 256) {
    float v = x[(long long)i * C + c];
    s += v; s2 += v * v;
  }
  __shared__ float sh[256], sh2[256];
  sh[threadIdx.x] = s; sh2[threadIdx.x] = s2;
  __syncthreads();
  for (int o = 128; o > 0; o >>= 1) {
    if (threadIdx.x < o) { sh[threadIdx.x] += sh[threadIdx.x + o];
                           sh2[threadIdx.x] += sh2[threadIdx.x + o]; }
    __syncthreads();
  }
  if (threadIdx.x == 0) {
    float* mv = MV + (long long)blockIdx.z * mvSB;
    const float m = sh[0] / (float)N;
    mv[c]     = m;
    mv[C + c] = sh2[0] / (float)N - m * m;
  }
}

// BN apply (batch stats) + LeakyReLU(0.2), in place
__global__ __launch_bounds__(256) void k_bn_act(
    float* __restrict__ X, const float* __restrict__ MV,
    const float* __restrict__ G, const float* __restrict__ Be,
    int N, int C, long long xSB, long long mvSB)
{
  const long long i = (long long)blockIdx.x * 256 + threadIdx.x;
  if (i >= (long long)N * C) return;
  const int c = (int)(i % C);
  float* x = X + (long long)blockIdx.z * xSB;
  const float* mv = MV + (long long)blockIdx.z * mvSB;
  float r = (x[i] - mv[c]) * rsqrtf(mv[C + c] + 1e-5f) * G[c] + Be[c];
  x[i] = (r >= 0.f) ? r : 0.2f * r;
}

// Mean pool over 7 fine neighbors
__global__ __launch_bounds__(256) void k_pool(
    const float* __restrict__ X, const int* __restrict__ nbr,
    float* __restrict__ Out, int Nc, int C, long long xSB, long long oSB)
{
  const long long i = (long long)blockIdx.x * 256 + threadIdx.x;
  if (i >= (long long)Nc * C) return;
  const int v = (int)(i / C), c = (int)(i % C);
  const float* x = X + (long long)blockIdx.z * xSB;
  float s = 0.f;
#pragma unroll
  for (int j = 0; j < 7; ++j)
    s += x[(long long)nbr[(long long)v * 7 + j] * C + c];
  (Out + (long long)blockIdx.z * oSB)[(long long)v * C + c] = s * (1.f / 7.f);
}

// Upconv scatter (replicates reference's flat reshape(-1,C,2).mean(2) exactly)
__global__ __launch_bounds__(256) void k_upscatter(
    const float* __restrict__ Y, const int* __restrict__ top,
    const int* __restrict__ dwn, float* __restrict__ Dst,
    int Nc, int Nf, int C, int dstRow, long long ySB, long long dSB)
{
  const long long i = (long long)blockIdx.x * 256 + threadIdx.x;
  if (i >= (long long)Nf * C) return;
  const int r = (int)(i / C), c = (int)(i % C);
  const float* y = Y + (long long)blockIdx.z * ySB;
  float val;
  if (r < Nc) {
    val = y[(long long)top[r] * C + c];
  } else {
    const int e  = r - Nc;
    const int hi = (2 * c >= C) ? 1 : 0;
    const int row = dwn[2 * e + hi];
    const int cc  = 2 * c - (hi ? C : 0);
    val = 0.5f * (y[(long long)row * C + cc] + y[(long long)row * C + cc + 1]);
  }
  (Dst + (long long)blockIdx.z * dSB)[(long long)r * dstRow + c] = val;
}

// Channel-block copy into a concat buffer
__global__ __launch_bounds__(256) void k_copych(
    const float* __restrict__ Src, float* __restrict__ Dst,
    int N, int Csrc, int dstRow, int chOff, int chOffPerB,
    long long sSB, long long dSB)
{
  const long long i = (long long)blockIdx.x * 256 + threadIdx.x;
  if (i >= (long long)N * Csrc) return;
  const int r = (int)(i / Csrc), c = (int)(i % Csrc);
  const float* s = Src + (long long)blockIdx.z * sSB;
  float*       d = Dst + (long long)blockIdx.z * dSB;
  d[(long long)r * dstRow + chOff + chOffPerB * (int)blockIdx.z + c] = s[i];
}

// ---------------------------------------------------------------------------
// Host orchestration
// ---------------------------------------------------------------------------
extern "C" void kernel_launch(void* const* d_in, const int* in_sizes, int n_in,
                              void* d_out, int out_size, void* d_ws, size_t ws_size,
                              hipStream_t stream)
{
  (void)in_sizes; (void)n_in; (void)out_size; (void)ws_size;

  static const int kNodes[5] = {40962, 10242, 2562, 642, 162};
  static const int kCh[6]    = {2, 8, 16, 32, 64, 128};

  auto F = [&](int i) -> const float* { return (const float*)d_in[i]; };
  auto I = [&](int i) -> const int*   { return (const int*)d_in[i]; };

  // ---- Input pointer map (setup_inputs insertion order, flattened) ----
  const float* X  = F(0);            // [8, 40962, 2]
  const float* X2 = F(1);            // [40962, 2]
  const float* dnp[5][2][4];         // [level][c1/c2][W,b,g,be]
  for (int i = 0; i < 5; ++i)
    for (int cc = 0; cc < 2; ++cc)
      for (int q = 0; q < 4; ++q)
        dnp[i][cc][q] = F(2 + 8 * i + 4 * cc + q);
  const float* upP[4][10];           // up_parc: upW,upb,c1W,c1b,c1g,c1be,c2W,c2b,c2g,c2be
  const float* upR[4][10];           // up_reg : same layout
  for (int j = 0; j < 4; ++j)
    for (int q = 0; q < 10; ++q) { upP[j][q] = F(42 + 10 * j + q); upR[j][q] = F(82 + 10 * j + q); }
  const float* parcW = F(122); const float* parcb = F(123);
  const float* regW  = F(124); const float* regb  = F(125);
  const int* nbrI[5]; for (int t = 0; t < 5; ++t) nbrI[t] = I(126 + t);
  const int* topI[4]; for (int t = 0; t < 4; ++t) topI[t] = I(131 + t);
  const int* dwnI[4]; for (int t = 0; t < 4; ++t) dwnI[t] = I(135 + t);

  float* ws   = (float*)d_ws;
  float* outF = (float*)d_out;

  // ---- Workspace layout (float offsets; regions 64B aligned) ----
  long long xsOff[5];
  {
    long long o = 0;
    for (int i = 0; i < 5; ++i) { xsOff[i] = o; o += (long long)kNodes[i] * kCh[i + 1]; }
  }
  const long long XS_PATH  = 635376;
  const long long WS_XS    = 0;
  const long long TMPA_S   = 720896;  const long long WS_TMPA = WS_XS   + 9 * XS_PATH;
  const long long TMPB_S   = 335872;  const long long WS_TMPB = WS_TMPA + 9 * TMPA_S;
  const long long TMPC_S   = 602112;  const long long WS_TMPC = WS_TMPB + 9 * TMPB_S;
  const long long TMPY_S   = 335872;  const long long WS_TMPY = WS_TMPC + 9 * TMPC_S;
  const long long MV_S     = 256;     const long long WS_MV   = WS_TMPY + 9 * TMPY_S;
  long long featOff[6];
  long long featEnd;
  {
    static const int featN[6] = {40962, 40962, 10242, 2562, 642, 162};
    long long o = WS_MV + 9 * MV_S;
    for (int i = 0; i < 6; ++i) { featOff[i] = o; o += (long long)featN[i] * kCh[i] * 9; }
    featEnd = (o + 15) & ~15ll;
  }
  const long long APACK_S  = 5243136;                 // bf16 elems per batch (=40962*128)
  const long long WS_APACK = featEnd;                 // bf16 region, 9*APACK_S elems
  const long long WS_WPACK = WS_APACK + (9 * APACK_S) / 2;   // bf16 region, 4M elems
  bf16_t* const APACK = (bf16_t*)(ws + WS_APACK);
  bf16_t* const WPACK = (bf16_t*)(ws + WS_WPACK);
  long long wcur = 0;                                 // deterministic weight cursor

  auto magic = [](int d) -> unsigned long long {
    return ((1ull << 36) + (unsigned)d - 1) / (unsigned)d;
  };
  auto ew_grid = [](long long tot, int B) { return dim3((unsigned)((tot + 255) / 256), 1, B); };

  // ---- pack(A[,gather]) + pack(W) + dense bf16 WMMA GEMM ----
  auto gemmX = [&](const float* Xp, long long xsb, const int* nb, int Cin,
                   const float* Wt, const float* bias, float* Op, long long osb,
                   int N, int Cout, int K, int orow, int B) {
    const int Kp = (K + 31) & ~31;
    const unsigned long long Mkp = magic(Kp);
    const long long atot = (long long)N * Kp;
    if (nb)
      k_pack_gather<<<ew_grid(atot, B), 256, 0, stream>>>(
          Xp, nb, APACK, N, Cin, K, Kp, Mkp, magic(Cin), xsb, APACK_S);
    else
      k_pack_dense<<<ew_grid(atot, B), 256, 0, stream>>>(
          Xp, APACK, N, K, Kp, Mkp, xsb, APACK_S);
    bf16_t* Wp = WPACK + wcur;
    const long long wtot = (long long)Cout * Kp;
    wcur += (wtot + 15) & ~15ll;
    k_pack_dense<<<ew_grid(wtot, 1), 256, 0, stream>>>(Wt, Wp, Cout, K, Kp, Mkp, 0, 0);
    dim3 g((N + 63) / 64, (Cout + 15) / 16, B);
    k_gemm_bf16<<<g, dim3(128), 0, stream>>>(APACK, Wp, bias, Op, N, Cout, Kp, orow, APACK_S, osb);
  };

  auto bn = [&](float* Xp, long long xsb, const float* g_, const float* be_,
                int N, int C, int B, int pb) {
    float* mv = ws + WS_MV + (long long)pb * MV_S;
    k_bn_stats<<<dim3(C, 1, B), 256, 0, stream>>>(Xp, mv, N, C, xsb, MV_S);
    k_bn_act<<<ew_grid((long long)N * C, B), 256, 0, stream>>>(
        Xp, mv, g_, be_, N, C, xsb, MV_S);
  };

  // ---- Encoder (down path), batched over B paths, temp slot base pb ----
  auto down_path = [&](const float* in0, long long in0sb, int pb, int B) {
    float* tA = ws + WS_TMPA + (long long)pb * TMPA_S;
    float* tB = ws + WS_TMPB + (long long)pb * TMPB_S;
    for (int i = 0; i < 5; ++i) {
      const int Nn = kNodes[i], Cin = kCh[i], Cout = kCh[i + 1];
      const float* src; long long ssb;
      if (i == 0) { src = in0; ssb = in0sb; }
      else {
        float* xsPrev = ws + WS_XS + (long long)pb * XS_PATH + xsOff[i - 1];
        k_pool<<<ew_grid((long long)Nn * Cin, B), 256, 0, stream>>>(
            xsPrev, nbrI[i - 1], tB, Nn, Cin, XS_PATH, TMPB_S);
        src = tB; ssb = TMPB_S;
      }
      gemmX(src, ssb, nbrI[i], Cin, dnp[i][0][0], dnp[i][0][1], tA, TMPA_S,
            Nn, Cout, 7 * Cin, Cout, B);
      bn(tA, TMPA_S, dnp[i][0][2], dnp[i][0][3], Nn, Cout, B, pb);
      float* xsCur = ws + WS_XS + (long long)pb * XS_PATH + xsOff[i];
      gemmX(tA, TMPA_S, nbrI[i], Cout, dnp[i][1][0], dnp[i][1][1], xsCur, XS_PATH,
            Nn, Cout, 7 * Cout, Cout, B);
      bn(xsCur, XS_PATH, dnp[i][1][2], dnp[i][1][3], Nn, Cout, B, pb);
    }
  };

  // ---- Parcellation decoder, batched ----
  auto up_parc = [&](int pb, int B, float* outP, long long outSB) {
    float* tA = ws + WS_TMPA + (long long)pb * TMPA_S;
    float* tB = ws + WS_TMPB + (long long)pb * TMPB_S;
    float* tC = ws + WS_TMPC + (long long)pb * TMPC_S;
    float* tY = ws + WS_TMPY + (long long)pb * TMPY_S;
    const float* y = ws + WS_XS + (long long)pb * XS_PATH + xsOff[4];
    long long ysb = XS_PATH; int Cy = 128;
    for (int j = 0; j < 4; ++j) {
      const int t = 3 - j, cout = kCh[4 - j];
      const int Nf = kNodes[t], Nc = kNodes[t + 1];
      gemmX(y, ysb, nullptr, Cy, upP[j][0], upP[j][1], tC, TMPC_S,
            Nc, 7 * cout, Cy, 7 * cout, B);
      k_upscatter<<<ew_grid((long long)Nf * cout, B), 256, 0, stream>>>(
          tC, topI[t], dwnI[t], tA, Nc, Nf, cout, 2 * cout, TMPC_S, TMPA_S);
      const float* skip = ws + WS_XS + (long long)pb * XS_PATH + xsOff[3 - j];
      k_copych<<<ew_grid((long long)Nf * cout, B), 256, 0, stream>>>(
          skip, tA, Nf, cout, 2 * cout, cout, 0, XS_PATH, TMPA_S);
      gemmX(tA, TMPA_S, nbrI[t], 2 * cout, upP[j][2], upP[j][3], tB, TMPB_S,
            Nf, cout, 14 * cout, cout, B);
      bn(tB, TMPB_S, upP[j][4], upP[j][5], Nf, cout, B, pb);
      gemmX(tB, TMPB_S, nbrI[t], cout, upP[j][6], upP[j][7], tY, TMPY_S,
            Nf, cout, 7 * cout, cout, B);
      bn(tY, TMPY_S, upP[j][8], upP[j][9], Nf, cout, B, pb);
      y = tY; ysb = TMPY_S; Cy = cout;
    }
    gemmX(y, ysb, nullptr, 8, parcW, parcb, outP, outSB, kNodes[0], 36, 8, 36, B);
  };

  // ---- Run the 8 scans (batched) and the atlas path ----
  down_path(X, (long long)kNodes[0] * 2, 0, 8);
  up_parc(0, 8, outF, (long long)kNodes[0] * 36);
  down_path(X2, 0, 8, 1);
  up_parc(8, 1, outF + (long long)8 * kNodes[0] * 36, 0);

  // ---- Joint features: feats[i] = [atlas | scan0..scan7] along channels ----
  {
    static const int featN[6] = {40962, 40962, 10242, 2562, 642, 162};
    for (int i = 0; i < 6; ++i) {
      const int Nn = featN[i], C = kCh[i], rs = 9 * C;
      float* dst = ws + featOff[i];
      const float* srcA = (i == 0) ? X2 : ws + WS_XS + 8 * XS_PATH + xsOff[i - 1];
      k_copych<<<ew_grid((long long)Nn * C, 1), 256, 0, stream>>>(
          srcA, dst, Nn, C, rs, 0, 0, 0, 0);
      const float* srcS = (i == 0) ? X : ws + WS_XS + xsOff[i - 1];
      long long ssb = (i == 0) ? (long long)kNodes[0] * 2 : XS_PATH;
      k_copych<<<ew_grid((long long)Nn * C, 8), 256, 0, stream>>>(
          srcS, dst, Nn, C, rs, C, C, ssb, 0);
    }
  }

  // ---- Registration decoder (single joint path) ----
  {
    float* tA = ws + WS_TMPA; float* tB = ws + WS_TMPB;
    float* tC = ws + WS_TMPC; float* tY = ws + WS_TMPY;
    const float* u = ws + featOff[5]; int Cu = 1152;
    for (int j = 0; j < 4; ++j) {
      const int t = 3 - j, cout = kCh[4 - j];
      const int Nf = kNodes[t], Nc = kNodes[t + 1];
      gemmX(u, 0, nullptr, Cu, upR[j][0], upR[j][1], tC, 0, Nc, 7 * cout, Cu, 7 * cout, 1);
      k_upscatter<<<ew_grid((long long)Nf * cout, 1), 256, 0, stream>>>(
          tC, topI[t], dwnI[t], tA, Nc, Nf, cout, 10 * cout, 0, 0);
      k_copych<<<ew_grid((long long)Nf * 9 * cout, 1), 256, 0, stream>>>(
          ws + featOff[4 - j], tA, Nf, 9 * cout, 10 * cout, cout, 0, 0, 0);
      gemmX(tA, 0, nbrI[t], 10 * cout, upR[j][2], upR[j][3], tB, 0,
            Nf, cout, 70 * cout, cout, 1);
      bn(tB, 0, upR[j][4], upR[j][5], Nf, cout, 1, 0);
      gemmX(tB, 0, nbrI[t], cout, upR[j][6], upR[j][7], tY, 0,
            Nf, cout, 7 * cout, cout, 1);
      bn(tY, 0, upR[j][8], upR[j][9], Nf, cout, 1, 0);
      u = tY; Cu = cout;
    }
    gemmX(u, 0, nullptr, 8, regW, regb, outF + (long long)9 * kNodes[0] * 36, 0,
          kNodes[0], 16, 8, 16, 1);
  }
}
